// KGMC_17789754540837
// MI455X (gfx1250) — compile-verified, matrix operand order checked by reading, not run on hardware
//
#include <hip/hip_runtime.h>
#include <hip/hip_bf16.h>
#include <math.h>

typedef __attribute__((ext_vector_type(16))) __bf16 v16bf;
typedef __attribute__((ext_vector_type(8)))  float  v8f;

#define HID  32
#define OUTW 96   // [V_b0 | V_b1 | loop] concatenated outputs

// ---------------------------------------------------------------- zero
__global__ void zero_f32(float* __restrict__ p, int n) {
    int i = blockIdx.x * blockDim.x + threadIdx.x;
    if (i < n) p[i] = 0.f;
}

// ------------------------------------------------- weight prepack (bf16)
// Pack W = [V0 | V1 | loopw]  (D x 96, f32) into per-lane WMMA B-fragment
// order (ISA 16-bit 32x16 B layout): element index
//   ((kt*6 + nt)*32 + lane)*16 + j
// lane: half = lane>>4 selects K-halves {0..7,16..23} vs {8..15,24..31},
//       col  = lane&15 is the N column within the 16-wide tile.
__global__ void prepack_w(const float* __restrict__ V,
                          const float* __restrict__ loopw,
                          __bf16* __restrict__ wpack, int D) {
    int t = blockIdx.x * blockDim.x + threadIdx.x;
    int KT = D >> 5;
    int total = KT * 6 * 32 * 16;
    if (t >= total) return;
    int j    = t & 15;
    int lane = (t >> 4) & 31;
    int ntkt = t >> 9;
    int nt   = ntkt % 6;
    int kt   = ntkt / 6;
    int half = lane >> 4;
    int col  = lane & 15;
    int k = kt * 32 + ((j < 8) ? (half * 8 + j) : (16 + half * 8 + (j - 8)));
    int n = nt * 16 + col;
    float v;
    if (n < 64) {                      // basis matrices V[b][k][o], [B,D,32]
        int b = n >> 5, o = n & 31;
        v = V[((size_t)b * D + k) * HID + o];
    } else {                           // self-loop weight [D,32]
        v = loopw[(size_t)k * HID + (n - 64)];
    }
    wpack[t] = (__bf16)v;
}

// ---------------------------------------------- dense transform via WMMA
// T[N x 96] = X[N x D] @ Wpack(D x 96), bf16 inputs, f32 accumulate.
// One wave per 16-row tile; 6 N-tiles of 16; KT = D/32 K-tiles.
template <int KT>
__global__ void rgcn_dense(const float* __restrict__ X,
                           const __bf16* __restrict__ wpack,
                           float* __restrict__ T, int nrows) {
    const int D = KT * 32;
    int lane = threadIdx.x & 31;
    int wave = threadIdx.x >> 5;
    int tile = blockIdx.x * 8 + wave;
    int row0 = tile * 16;
    if (row0 >= nrows) return;        // uniform per wave: EXEC stays all-1s
    int half = lane >> 4;
    int r    = lane & 15;

    // A fragments: lane holds row (row0+r), K = half*8+{0..7} and 16+half*8+{0..7}
    v16bf a[KT];
    const float* xr = X + (size_t)(row0 + r) * D + half * 8;
#pragma unroll
    for (int kt = 0; kt < KT; ++kt) {
        const float4* p = reinterpret_cast<const float4*>(xr + kt * 32);
        const float4* q = reinterpret_cast<const float4*>(xr + kt * 32 + 16);
        float4 f0 = p[0], f1 = p[1], f2 = q[0], f3 = q[1];
        v16bf av;
        av[0]  = (__bf16)f0.x; av[1]  = (__bf16)f0.y;
        av[2]  = (__bf16)f0.z; av[3]  = (__bf16)f0.w;
        av[4]  = (__bf16)f1.x; av[5]  = (__bf16)f1.y;
        av[6]  = (__bf16)f1.z; av[7]  = (__bf16)f1.w;
        av[8]  = (__bf16)f2.x; av[9]  = (__bf16)f2.y;
        av[10] = (__bf16)f2.z; av[11] = (__bf16)f2.w;
        av[12] = (__bf16)f3.x; av[13] = (__bf16)f3.y;
        av[14] = (__bf16)f3.z; av[15] = (__bf16)f3.w;
        a[kt] = av;
    }

    const v16bf* wp = reinterpret_cast<const v16bf*>(wpack);
#pragma unroll
    for (int nt = 0; nt < 6; ++nt) {
        v8f acc = {};
#pragma unroll
        for (int kt = 0; kt < KT; ++kt) {
            v16bf b = wp[(kt * 6 + nt) * 32 + lane];
            acc = __builtin_amdgcn_wmma_f32_16x16x32_bf16(
                /*neg_a=*/false, a[kt], /*neg_b=*/false, b,
                /*c_mod=*/(short)0, acc, /*reuse_a=*/false, /*reuse_b=*/false);
        }
        // C/D layout: lane (half,r) holds rows M = half*8 + v, col N = r
        float* out = T + (size_t)(row0 + half * 8) * OUTW + nt * 16 + r;
#pragma unroll
        for (int v = 0; v < 8; ++v) out[(size_t)v * OUTW] = acc[v];
    }
}

// ------------------------------------------------ edge message + scatter
// One wave per edge; lane = output feature (HID==32==wave32).
__global__ void edge_agg(const float* __restrict__ T,
                         const float* __restrict__ comp,
                         const float* __restrict__ mask,
                         const int* __restrict__ etype,
                         const int* __restrict__ src,
                         const int* __restrict__ dst,
                         float* __restrict__ agg, int E) {
    int lane = threadIdx.x & 31;
    int e = (blockIdx.x * blockDim.x + threadIdx.x) >> 5;
    if (e >= E) return;
    int t = etype[e], s = src[e], d = dst[e];
    float m  = mask[e];
    float c0 = comp[t * 2 + 0] * m;
    float c1 = comp[t * 2 + 1] * m;
    const float* ts = T + (size_t)s * OUTW;          // L2-resident gather
    float v = c0 * ts[lane] + c1 * ts[32 + lane];
    unsafeAtomicAdd(agg + (size_t)d * HID + lane, v); // global_atomic_add_f32
}

// ----------------------------------------------------------- activation
__global__ void finalize_h(const float* __restrict__ agg,
                           const float* __restrict__ T,
                           const float* __restrict__ bias,
                           float* __restrict__ h, int n_nodes) {
    int i = blockIdx.x * blockDim.x + threadIdx.x;
    if (i >= n_nodes * HID) return;
    int j = i & 31;
    int node = i >> 5;
    float v = agg[i] + T[(size_t)node * OUTW + 64 + j] + bias[j];
    h[i] = tanhf(v);
}

// ------------------------------------------------------------- MLP head
// One block (128 threads) per pair; feats staged in LDS.
__global__ void head_mlp(const float* __restrict__ h1,
                         const float* __restrict__ h2,
                         const float* __restrict__ h3,
                         const int* __restrict__ users,
                         const int* __restrict__ items,
                         const float* __restrict__ w1,
                         const float* __restrict__ bw1,
                         const float* __restrict__ w2,
                         const float* __restrict__ bw2,
                         float* __restrict__ out) {
    __shared__ float feats[192];
    __shared__ float red[128];
    int p = blockIdx.x;
    int tid = threadIdx.x;
    for (int t = tid; t < 192; t += 128) {
        int side = t / 96;                 // 0 = user, 1 = item
        int q = t % 96;
        int node = side ? items[p] : users[p];
        const float* hh = (q < 32) ? h1 : (q < 64) ? h2 : h3;
        feats[t] = hh[(size_t)node * HID + (q & 31)];
    }
    __syncthreads();
    float acc = bw1[tid];
    for (int k = 0; k < 192; ++k) acc += feats[k] * w1[k * 128 + tid];
    acc = fmaxf(acc, 0.f);
    red[tid] = acc * w2[tid];
    __syncthreads();
    for (int s = 64; s > 0; s >>= 1) {
        if (tid < s) red[tid] += red[tid + s];
        __syncthreads();
    }
    if (tid == 0) out[p] = red[0] + bw2[0];
}

// ---------------------------------------------------------------- launch
extern "C" void kernel_launch(void* const* d_in, const int* in_sizes, int n_in,
                              void* d_out, int out_size, void* d_ws, size_t ws_size,
                              hipStream_t stream) {
    const float* x         = (const float*)d_in[0];
    const float* edge_mask = (const float*)d_in[1];
    const int*   etype     = (const int*)d_in[2];
    const int*   src       = (const int*)d_in[3];
    const int*   dst       = (const int*)d_in[4];
    const int*   users     = (const int*)d_in[5];
    const int*   items     = (const int*)d_in[6];
    const float* V1    = (const float*)d_in[7];
    const float* comp1 = (const float*)d_in[8];
    const float* loop1 = (const float*)d_in[9];
    const float* b1    = (const float*)d_in[10];
    const float* V2    = (const float*)d_in[11];
    const float* comp2 = (const float*)d_in[12];
    const float* loop2 = (const float*)d_in[13];
    const float* b2    = (const float*)d_in[14];
    const float* V3    = (const float*)d_in[15];
    const float* comp3 = (const float*)d_in[16];
    const float* loop3 = (const float*)d_in[17];
    const float* b3    = (const float*)d_in[18];
    const float* w1    = (const float*)d_in[19];
    const float* bw1   = (const float*)d_in[20];
    const float* w2    = (const float*)d_in[21];
    const float* bw2   = (const float*)d_in[22];
    float* out = (float*)d_out;

    const int N = in_sizes[0] / 128;   // 100000 (divisible by 16)
    const int E = in_sizes[2];         // 1600000
    const int P = in_sizes[5];         // 4096

    // workspace layout (all chunks multiples of 256B; base is 256B aligned)
    char* ws = (char*)d_ws;
    float* T    = (float*)(ws);                                   // N*96
    float* agg  = T  + (size_t)N * OUTW;                          // N*32
    float* h1   = agg + (size_t)N * HID;                          // N*32
    float* h2   = h1  + (size_t)N * HID;                          // N*32
    float* h3   = h2  + (size_t)N * HID;                          // N*32
    __bf16* wpack = (__bf16*)(h3 + (size_t)N * HID);              // <=12288 bf16

    const int tiles   = N / 16;
    const int dblocks = (tiles + 7) / 8;
    const int nThr    = N * HID;
    const int eThr    = E * 32;

    // ---- layer 1 (D = 128) ----
    prepack_w<<<(4 * 6 * 32 * 16 + 255) / 256, 256, 0, stream>>>(V1, loop1, wpack, 128);
    zero_f32<<<(nThr + 255) / 256, 256, 0, stream>>>(agg, nThr);
    rgcn_dense<4><<<dblocks, 256, 0, stream>>>(x, wpack, T, N);
    edge_agg<<<(eThr + 255) / 256, 256, 0, stream>>>(T, comp1, edge_mask, etype, src, dst, agg, E);
    finalize_h<<<(nThr + 255) / 256, 256, 0, stream>>>(agg, T, b1, h1, N);

    // ---- layer 2 (D = 32) ----
    prepack_w<<<(1 * 6 * 32 * 16 + 255) / 256, 256, 0, stream>>>(V2, loop2, wpack, 32);
    zero_f32<<<(nThr + 255) / 256, 256, 0, stream>>>(agg, nThr);
    rgcn_dense<1><<<dblocks, 256, 0, stream>>>(h1, wpack, T, N);
    edge_agg<<<(eThr + 255) / 256, 256, 0, stream>>>(T, comp2, edge_mask, etype, src, dst, agg, E);
    finalize_h<<<(nThr + 255) / 256, 256, 0, stream>>>(agg, T, b2, h2, N);

    // ---- layer 3 (D = 32) ----
    prepack_w<<<(1 * 6 * 32 * 16 + 255) / 256, 256, 0, stream>>>(V3, loop3, wpack, 32);
    zero_f32<<<(nThr + 255) / 256, 256, 0, stream>>>(agg, nThr);
    rgcn_dense<1><<<dblocks, 256, 0, stream>>>(h2, wpack, T, N);
    edge_agg<<<(eThr + 255) / 256, 256, 0, stream>>>(T, comp3, edge_mask, etype, src, dst, agg, E);
    finalize_h<<<(nThr + 255) / 256, 256, 0, stream>>>(agg, T, b3, h3, N);

    // ---- MLP head ----
    head_mlp<<<P, 128, 0, stream>>>(h1, h2, h3, users, items, w1, bw1, w2, bw2, out);
}
